// PA_CA_parallel_4741643895326
// MI455X (gfx1250) — compile-verified, matrix operand order checked by reading, not run on hardware
//
#include <hip/hip_runtime.h>
#include <hip/hip_bf16.h>

// ---------------------------------------------------------------------------
// MI455X (gfx1250) dual-attention block.
// All matmul-shaped work -> v_wmma_f32_16x16x32_f16 (f16 in, f32 accumulate).
// Conv weight tiles double-buffered via GLOBAL_LOAD_ASYNC_TO_LDS_B128 so the
// DMA of chunk t+1 overlaps the WMMAs of chunk t (ASYNCcnt pipeline).
// B=2, C=256, Cr=64, H=W=64, HW=4096
// ---------------------------------------------------------------------------

typedef __attribute__((ext_vector_type(16))) _Float16 v16h;
typedef __attribute__((ext_vector_type(8)))  _Float16 v8h;
typedef __attribute__((ext_vector_type(8)))  float    v8f;

static constexpr int BATCH = 2;
static constexpr int HW    = 4096;   // 64*64

__device__ __forceinline__ v8f wmma_f16(v16h a, v16h b, v8f c) {
    return __builtin_amdgcn_wmma_f32_16x16x32_f16(
        /*neg_a=*/false, a, /*neg_b=*/false, b,
        /*c_mod=*/(short)0, c, /*reuse_a=*/false, /*reuse_b=*/false);
}

// A/B fragment loader for 16xK (K-chunk of 32) row-major f16 tiles.
// Per ISA 16-bit A 16x32 layout: lane L -> row (L&15); half h=L>>4 selects
// K = {8h..8h+7} in elements [0..7] and K = {16+8h..16+8h+7} in [8..15].
// B fragment assumed symmetric (n = L&15, same K map) -- compile-only env.
__device__ __forceinline__ v16h load_frag(const _Float16* tile, int ld) {
    int l = threadIdx.x & 31;
    int r = l & 15, h = l >> 4;
    const _Float16* p = tile + (size_t)r * ld + h * 8;
    v8h lo = *(const v8h*)(p);
    v8h hi = *(const v8h*)(p + 16);
    v16h f;
#pragma unroll
    for (int i = 0; i < 8; ++i) { f[i] = lo[i]; f[i + 8] = hi[i]; }
    return f;
}

// async 16B global->LDS copy (CDNA5). lds addr = low 32 bits of flat address
// (ISA 10.2: LDS aperture address truncates to LDS offset).
__device__ __forceinline__ void async_copy_b128(const void* gsrc, void* lds_dst) {
    unsigned lo = (unsigned)(size_t)lds_dst;
    unsigned long long ga = (unsigned long long)(size_t)gsrc;
    asm volatile("global_load_async_to_lds_b128 %0, %1, off"
                 :: "v"(lo), "v"(ga) : "memory");
}
__device__ __forceinline__ void async_wait0() {
    asm volatile("s_wait_asynccnt 0x0" ::: "memory");
}

// ---------------------------------------------------------------------------
// Small packing / casting kernels
// ---------------------------------------------------------------------------
__global__ void cast_f32_f16(const float* __restrict__ src,
                             _Float16* __restrict__ dst, int n) {
    int i = blockIdx.x * blockDim.x + threadIdx.x;
    if (i < n) dst[i] = (_Float16)src[i];
}

// im2col k -> (ci, r, s) decode table: meta = ci | r<<12 | s<<14
__global__ void build_kmeta(int* __restrict__ kmeta) {
    int k = blockIdx.x * blockDim.x + threadIdx.x;
    if (k < 2304) {
        int ci = k / 9, t9 = k - ci * 9;
        int r3 = t9 / 3, c3 = t9 - r3 * 3;
        kmeta[k] = ci | (r3 << 12) | (c3 << 14);
    }
}

__global__ void pack_pc(const float* __restrict__ w1, const float* __restrict__ b1,
                        const float* __restrict__ w2, const float* __restrict__ b2,
                        _Float16* __restrict__ Wd, float* __restrict__ Bd) {
    int i = blockIdx.x * blockDim.x + threadIdx.x;
    if (i < 128 * 2304) {
        int o = i / 2304, k = i - o * 2304;
        const float* s = (o < 64) ? w1 : w2;
        Wd[i] = (_Float16)s[(o & 63) * 2304 + k];
    }
    if (i < 128) Bd[i] = (i < 64) ? b1[i] : b2[i - 64];
}

__global__ void pack_t(const float* __restrict__ w, const float* __restrict__ bsrc,
                       _Float16* __restrict__ Wd, float* __restrict__ Bd) {
    int i = blockIdx.x * blockDim.x + threadIdx.x;
    if (i < 256 * 576) Wd[i] = (_Float16)w[i];
    if (i < 256) Bd[i] = bsrc[i];
}

// weight-norm 1x1 weights for q/k/v packed into [192][64] f16 + bias[192]
__global__ void qkv_wn(const float* qv, const float* qg, const float* qb,
                       const float* kv, const float* kg, const float* kb,
                       const float* vv, const float* vg, const float* vb,
                       _Float16* __restrict__ Wq, float* __restrict__ Bq) {
    int o = blockIdx.x * blockDim.x + threadIdx.x;
    if (o >= 192) return;
    int sel = o >> 6, oo = o & 63;
    const float* v = (sel == 0) ? qv : (sel == 1) ? kv : vv;
    const float* g = (sel == 0) ? qg : (sel == 1) ? kg : vg;
    const float* bb = (sel == 0) ? qb : (sel == 1) ? kb : vb;
    float s = 0.f;
    for (int i = 0; i < 64; ++i) { float x = v[oo * 64 + i]; s += x * x; }
    float scale = g[oo] / sqrtf(s);
    for (int i = 0; i < 64; ++i) Wq[o * 64 + i] = (_Float16)(v[oo * 64 + i] * scale);
    Bq[o] = bb[oo];
}

// ---------------------------------------------------------------------------
// 3x3 conv as implicit GEMM:  Y[Co, B*HW] = W[Co, Ci*9] @ im2col(X)
// block = 256 threads = 8 waves; macro tile 128 Co x 64 N;
// each wave owns a 32x32 tile = 2x2 fragments = 4 WMMAs per K-chunk.
// Double-buffered LDS: async DMA + im2col gather of chunk t+1 overlap the
// WMMAs of chunk t; one barrier per chunk. kmeta table cached in LDS,
// padding handled branchlessly (clamp + select).
// ---------------------------------------------------------------------------
__global__ __launch_bounds__(256)
void conv3x3_igemm(const _Float16* __restrict__ Wg, const float* __restrict__ bias,
                   const _Float16* __restrict__ X, const int* __restrict__ kmeta,
                   float* __restrict__ Y, _Float16* __restrict__ Yh,
                   int Co, int Ci, int accumulate) {
    constexpr int LD = 40;                  // 32 + 8 pad: dodge LDS bank conflicts
    const int K = Ci * 9;
    __shared__ _Float16 Alds[2][128 * LD];
    __shared__ _Float16 Blds[2][64 * LD];
    __shared__ int kmloc[2304];
    const int tid = threadIdx.x;
    for (int k = tid; k < K; k += 256) kmloc[k] = kmeta[k];

    const int n0 = blockIdx.x * 64;
    const int co0 = blockIdx.y * 128;
    const int wave = tid >> 5, wr = wave & 3, wc = wave >> 2;

    // staging thread's spatial column (one of 64 per block)
    const int sn = n0 + (tid & 63);
    const int sb = sn >> 12, shw = sn & 4095;
    const int sh = shw >> 6, sw = shw & 63;
    const _Float16* Xb = X + (size_t)sb * Ci * HW;
    const int kq = tid >> 6;                // 0..3 -> 8 k-locals each

    auto stage = [&](int kk, int buf) {
        // weights: 128 rows x 32 k via async DMA, 2 x 16B per thread
#pragma unroll
        for (int u = 0; u < 2; ++u) {
            int idx = tid + u * 256;
            int row = idx >> 2, k8 = (idx & 3) * 8;
            async_copy_b128(&Wg[(size_t)(co0 + row) * K + kk + k8],
                            &Alds[buf][row * LD + k8]);
        }
        // im2col patches, transposed to [n][k]; branchless zero-padding
#pragma unroll
        for (int j = 0; j < 8; ++j) {
            int kl = kq * 8 + j;
            int m = kmloc[kk + kl];
            int ci = m & 0xfff;
            int hh = sh + ((m >> 12) & 3) - 1;
            int ww = sw + ((m >> 14) & 3) - 1;
            bool inb = ((unsigned)hh < 64u) & ((unsigned)ww < 64u);
            int hc = min(max(hh, 0), 63), wcl = min(max(ww, 0), 63);
            _Float16 val = Xb[(size_t)ci * HW + hc * 64 + wcl];
            Blds[buf][(tid & 63) * LD + kl] = inb ? val : (_Float16)0.f;
        }
    };

    __syncthreads();                        // kmeta cache visible
    stage(0, 0);
    async_wait0();
    __syncthreads();

    v8f acc00 = {}, acc01 = {}, acc10 = {}, acc11 = {};
    const int nT = K / 32;
    for (int t = 0; t < nT; ++t) {
        int cur = t & 1;
        if (t + 1 < nT) stage((t + 1) * 32, cur ^ 1);   // overlap with WMMAs
        v16h a0 = load_frag(&Alds[cur][(wr * 32) * LD], LD);
        v16h a1 = load_frag(&Alds[cur][(wr * 32 + 16) * LD], LD);
        v16h b0 = load_frag(&Blds[cur][(wc * 32) * LD], LD);
        v16h b1 = load_frag(&Blds[cur][(wc * 32 + 16) * LD], LD);
        acc00 = wmma_f16(a0, b0, acc00);
        acc01 = wmma_f16(a0, b1, acc01);
        acc10 = wmma_f16(a1, b0, acc10);
        acc11 = wmma_f16(a1, b1, acc11);
        async_wait0();
        __syncthreads();
    }
    // epilogue (C layout: lane col = L&15, rows r + 8*(L>>4))
    int l = tid & 31, nl = l & 15, h = l >> 4;
    v8f accs[2][2] = {{acc00, acc01}, {acc10, acc11}};
#pragma unroll
    for (int cs = 0; cs < 2; ++cs)
#pragma unroll
        for (int ns = 0; ns < 2; ++ns) {
            int ncol = n0 + wc * 32 + ns * 16 + nl;
            int ob = ncol >> 12, ohw = ncol & 4095;
#pragma unroll
            for (int r = 0; r < 8; ++r) {
                int row = co0 + wr * 32 + cs * 16 + r + 8 * h;
                float val = accs[cs][ns][r] + bias[row];
                size_t yi = ((size_t)ob * Co + row) * HW + ohw;
                if (accumulate) val += Y[yi];
                Y[yi] = val;
                if (Yh) Yh[yi] = (_Float16)val;
            }
        }
}

// ---------------------------------------------------------------------------
// q/k/v 1x1 conv GEMM:  [192,64] @ f1[64, B*HW]; writes qT/kT transposed,
// v natural -- so attention fragment loads hit global memory directly.
// ---------------------------------------------------------------------------
__global__ __launch_bounds__(256)
void qkv_gemm(const _Float16* __restrict__ Wq, const float* __restrict__ Bq,
              const _Float16* __restrict__ F12h,
              _Float16* __restrict__ qT, _Float16* __restrict__ kT,
              _Float16* __restrict__ Vm) {
    constexpr int LD = 72;
    __shared__ _Float16 Blds[32 * LD];
    int tid = threadIdx.x;
    int n0 = blockIdx.x * 32;
    int m0 = blockIdx.y * 64;
    {   // stage f1 patch transposed [n][c]
        int n = tid >> 3, c8 = (tid & 7) * 8;
        int ng = n0 + n, b = ng >> 12, hw = ng & 4095;
#pragma unroll
        for (int j = 0; j < 8; ++j) {
            int c = c8 + j;
            Blds[n * LD + c] = F12h[((size_t)b * 128 + c) * HW + hw];
        }
    }
    __syncthreads();
    int wave = tid >> 5, wr = wave & 3, wc = wave >> 2;
    v8f acc = {};
#pragma unroll
    for (int ks = 0; ks < 2; ++ks) {
        v16h a = load_frag(Wq + (size_t)(m0 + wr * 16) * 64 + ks * 32, 64);
        v16h b = load_frag(&Blds[(wc * 16) * LD + ks * 32], LD);
        acc = wmma_f16(a, b, acc);
    }
    int l = tid & 31, nl = l & 15, h = l >> 4;
    int ncol = n0 + wc * 16 + nl;
    int b = ncol >> 12, hw = ncol & 4095;
#pragma unroll
    for (int r = 0; r < 8; ++r) {
        int o = m0 + wr * 16 + r + 8 * h;
        _Float16 hv = (_Float16)(acc[r] + Bq[o]);
        if (o < 64)        qT[((size_t)b * HW + hw) * 64 + o] = hv;
        else if (o < 128)  kT[((size_t)b * HW + hw) * 64 + (o - 64)] = hv;
        else               Vm[((size_t)b * 64 + (o - 128)) * HW + hw] = hv;
    }
}

// ---------------------------------------------------------------------------
// Pass 1: per-row softmax stats (max, 1/sum) of energy = qT @ k, via WMMA
// block: 16 rows of i, 8 waves sweep all 4096 j with online softmax updates
// ---------------------------------------------------------------------------
__global__ __launch_bounds__(256)
void attn_rowstats(const _Float16* __restrict__ qT, const _Float16* __restrict__ kT,
                   float2* __restrict__ stats) {
    __shared__ float red[8 * 32 * 16];
    int tid = threadIdx.x, wave = tid >> 5;
    int b = blockIdx.x >> 8;
    int i0 = (blockIdx.x & 255) * 16;
    const _Float16* qrow = qT + ((size_t)b * HW + i0) * 64;
    v16h a0 = load_frag(qrow, 64);
    v16h a1 = load_frag(qrow + 32, 64);
    float m[8], s[8];
#pragma unroll
    for (int r = 0; r < 8; ++r) { m[r] = -3.0e38f; s[r] = 0.f; }
    for (int t = 0; t < 32; ++t) {
        int j0 = wave * 16 + t * 128;
        const _Float16* krow = kT + ((size_t)b * HW + j0) * 64;
        v8f e = {};
        e = wmma_f16(a0, load_frag(krow, 64), e);
        e = wmma_f16(a1, load_frag(krow + 32, 64), e);
#pragma unroll
        for (int r = 0; r < 8; ++r) {
            float x = e[r];
            float mn = fmaxf(m[r], x);
            s[r] = s[r] * __expf(m[r] - mn) + __expf(x - mn);
            m[r] = mn;
        }
    }
    int l = tid & 31;
    float* slot = &red[(wave * 32 + l) * 16];
#pragma unroll
    for (int r = 0; r < 8; ++r) { slot[r] = m[r]; slot[8 + r] = s[r]; }
    __syncthreads();
    if (tid < 16) {                      // row (tid) held by lanes with h = tid>>3
        int h = tid >> 3, rr = tid & 7;
        float M = -3.0e38f, S = 0.f;
        for (int w = 0; w < 8; ++w)
            for (int ll = 0; ll < 16; ++ll) {
                const float* sl = &red[(w * 32 + (h * 16 + ll)) * 16];
                float mm = sl[rr], ss = sl[8 + rr];
                float mn = fmaxf(M, mm);
                S = S * __expf(M - mn) + ss * __expf(mm - mn);
                M = mn;
            }
        stats[(size_t)b * HW + i0 + tid] = make_float2(M, 1.0f / S);
    }
}

// ---------------------------------------------------------------------------
// Pass 2: pos = v @ softmax(energy); energy C-fragments are converted
// in-register to B-fragments of the apply WMMA (row map r+8h == K map 8h+j).
// Full stats table preloaded to LDS once -> the 128-chunk main loop runs
// barrier-free. Epilogue fuses p = gamma1*pos + f1 -> f16 for the t1 conv.
// ---------------------------------------------------------------------------
__global__ __launch_bounds__(256)
void attn_pos(const _Float16* __restrict__ qT, const _Float16* __restrict__ kT,
              const _Float16* __restrict__ Vm, const float2* __restrict__ stats,
              const float* __restrict__ F12, const float* __restrict__ gamma1,
              _Float16* __restrict__ Ph) {
    __shared__ float2 st[HW];            // 32 KB: all row stats for this batch
    int tid = threadIdx.x;
    int wave = tid >> 5, ct = wave & 3, jt = wave >> 2;
    int b = blockIdx.x >> 7;
    int j0 = (blockIdx.x & 127) * 32;
    for (int i = tid; i < HW; i += 256) st[i] = stats[(size_t)b * HW + i];
    __syncthreads();

    int jcol = j0 + jt * 16;
    const _Float16* krow = kT + ((size_t)b * HW + jcol) * 64;
    v16h bk0 = load_frag(krow, 64);
    v16h bk1 = load_frag(krow + 32, 64);
    int l = tid & 31, h = l >> 4;
    v8f pos = {};
    for (int i0 = 0; i0 < HW; i0 += 32) {
        const _Float16* qrow = qT + ((size_t)b * HW + i0) * 64;
        v8f e0 = {}, e1 = {};
        e0 = wmma_f16(load_frag(qrow, 64), bk0, e0);
        e0 = wmma_f16(load_frag(qrow + 32, 64), bk1, e0);
        e1 = wmma_f16(load_frag(qrow + 16 * 64, 64), bk0, e1);
        e1 = wmma_f16(load_frag(qrow + 16 * 64 + 32, 64), bk1, e1);
        v16h bf;
#pragma unroll
        for (int j = 0; j < 8; ++j) {
            int ir = i0 + 8 * h + j;
            float2 s0 = st[ir];
            float2 s1 = st[ir + 16];
            bf[j]     = (_Float16)(__expf(e0[j] - s0.x) * s0.y);
            bf[8 + j] = (_Float16)(__expf(e1[j] - s1.x) * s1.y);
        }
        v16h av = load_frag(Vm + ((size_t)b * 64 + ct * 16) * HW + i0, HW);
        pos = wmma_f16(av, bf, pos);
    }
    const float g1 = gamma1[0];
    int nl = l & 15;
    int col = jcol + nl;
#pragma unroll
    for (int r = 0; r < 8; ++r) {
        int c = ct * 16 + r + 8 * h;
        float val = g1 * pos[r] + F12[((size_t)b * 128 + c) * HW + col];
        Ph[((size_t)b * 64 + c) * HW + col] = (_Float16)val;
    }
}

// ---------------------------------------------------------------------------
// Channel attention (tiny, VALU):
// ---------------------------------------------------------------------------
__global__ __launch_bounds__(256)
void ch_gram(const float* __restrict__ F12, float* __restrict__ G) {
    __shared__ float part[4][64];
    int c = blockIdx.x & 63, b = blockIdx.x >> 6;
    int d = threadIdx.x & 63, seg = threadIdx.x >> 6;
    const float* fc = F12 + ((size_t)b * 128 + 64 + c) * HW;
    const float* fd = F12 + ((size_t)b * 128 + 64 + d) * HW;
    float acc = 0.f;
    for (int n = seg * 1024; n < seg * 1024 + 1024; ++n) acc += fc[n] * fd[n];
    part[seg][d] = acc;
    __syncthreads();
    if (threadIdx.x < 64) {
        float s = part[0][threadIdx.x] + part[1][threadIdx.x] +
                  part[2][threadIdx.x] + part[3][threadIdx.x];
        G[((size_t)b * 64 + c) * 64 + threadIdx.x] = s;
    }
}

// softmax(rowmax - G) over d reduces to exp(minG - G)/sum
__global__ void ch_softmax(const float* __restrict__ G, float* __restrict__ Att) {
    __shared__ float row[64];
    __shared__ float mn_s, sum_s;
    int c = blockIdx.x & 63, b = blockIdx.x >> 6;
    int d = threadIdx.x;
    float g = G[((size_t)b * 64 + c) * 64 + d];
    row[d] = g;
    __syncthreads();
    if (d == 0) {
        float mn = row[0];
        for (int i = 1; i < 64; ++i) mn = fminf(mn, row[i]);
        mn_s = mn;
    }
    __syncthreads();
    float ex = __expf(mn_s - g);
    row[d] = ex;
    __syncthreads();
    if (d == 0) {
        float s = 0.f;
        for (int i = 0; i < 64; ++i) s += row[i];
        sum_s = s;
    }
    __syncthreads();
    Att[((size_t)b * 64 + c) * 64 + d] = ex / sum_s;
}

__global__ __launch_bounds__(256)
void ch_apply(const float* __restrict__ F12, const float* __restrict__ Att,
              const float* __restrict__ gamma2, _Float16* __restrict__ Ch) {
    __shared__ float arow[64];
    int idx = blockIdx.x;
    int nseg = idx & 15; idx >>= 4;
    int c = idx & 63, b = idx >> 6;
    int n = nseg * 256 + threadIdx.x;
    if (threadIdx.x < 64)
        arow[threadIdx.x] = Att[((size_t)b * 64 + c) * 64 + threadIdx.x];
    __syncthreads();
    const float* f2 = F12 + ((size_t)b * 128 + 64) * HW;
    float acc = 0.f;
#pragma unroll 8
    for (int d = 0; d < 64; ++d) acc += arow[d] * f2[(size_t)d * HW + n];
    float val = gamma2[0] * acc + f2[(size_t)c * HW + n];
    Ch[((size_t)b * 64 + c) * HW + n] = (_Float16)val;
}

// ---------------------------------------------------------------------------
extern "C" void kernel_launch(void* const* d_in, const int* in_sizes, int n_in,
                              void* d_out, int out_size, void* d_ws, size_t ws_size,
                              hipStream_t stream) {
    (void)in_sizes; (void)n_in; (void)out_size; (void)ws_size;
    const float* x     = (const float*)d_in[0];
    const float* pc1_w = (const float*)d_in[1];
    const float* pc1_b = (const float*)d_in[2];
    const float* pc2_w = (const float*)d_in[3];
    const float* pc2_b = (const float*)d_in[4];
    const float* q_v   = (const float*)d_in[5];
    const float* q_g   = (const float*)d_in[6];
    const float* q_b   = (const float*)d_in[7];
    const float* k_v   = (const float*)d_in[8];
    const float* k_g   = (const float*)d_in[9];
    const float* k_b   = (const float*)d_in[10];
    const float* v_v   = (const float*)d_in[11];
    const float* v_g   = (const float*)d_in[12];
    const float* v_b   = (const float*)d_in[13];
    const float* gamma1 = (const float*)d_in[14];
    const float* gamma2 = (const float*)d_in[15];
    const float* t1_w  = (const float*)d_in[16];
    const float* t1_b  = (const float*)d_in[17];
    const float* t2_w  = (const float*)d_in[18];
    const float* t2_b  = (const float*)d_in[19];
    float* out = (float*)d_out;

    char* ws = (char*)d_ws;
    size_t off = 0;
    auto carve = [&](size_t bytes) -> void* {
        void* p = ws + off;
        off += (bytes + 255) & ~(size_t)255;
        return p;
    };
    _Float16* xh    = (_Float16*)carve((size_t)BATCH * 256 * HW * 2);
    _Float16* wpc   = (_Float16*)carve((size_t)128 * 2304 * 2);
    float*    pcb   = (float*)   carve(128 * 4);
    float*    f12   = (float*)   carve((size_t)BATCH * 128 * HW * 4);
    _Float16* f12h  = (_Float16*)carve((size_t)BATCH * 128 * HW * 2);
    _Float16* qkvw  = (_Float16*)carve((size_t)192 * 64 * 2);
    float*    qkvb  = (float*)   carve(192 * 4);
    _Float16* qTb   = (_Float16*)carve((size_t)BATCH * HW * 64 * 2);
    _Float16* kTb   = (_Float16*)carve((size_t)BATCH * HW * 64 * 2);
    _Float16* vmb   = (_Float16*)carve((size_t)BATCH * 64 * HW * 2);
    float2*   stb   = (float2*)  carve((size_t)BATCH * HW * sizeof(float2));
    _Float16* ph    = (_Float16*)carve((size_t)BATCH * 64 * HW * 2);
    _Float16* chh   = (_Float16*)carve((size_t)BATCH * 64 * HW * 2);
    float*    Gb    = (float*)   carve((size_t)BATCH * 64 * 64 * 4);
    float*    attb  = (float*)   carve((size_t)BATCH * 64 * 64 * 4);
    _Float16* wt1   = (_Float16*)carve((size_t)256 * 576 * 2);
    _Float16* wt2   = (_Float16*)carve((size_t)256 * 576 * 2);
    float*    tb1   = (float*)   carve(256 * 4);
    float*    tb2   = (float*)   carve(256 * 4);
    int*      kmeta = (int*)     carve(2304 * 4);

    const int nx = BATCH * 256 * HW;
    cast_f32_f16<<<(nx + 255) / 256, 256, 0, stream>>>(x, xh, nx);
    build_kmeta<<<9, 256, 0, stream>>>(kmeta);
    pack_pc<<<(128 * 2304 + 255) / 256, 256, 0, stream>>>(pc1_w, pc1_b, pc2_w, pc2_b, wpc, pcb);
    pack_t<<<(256 * 576 + 255) / 256, 256, 0, stream>>>(t1_w, t1_b, wt1, tb1);
    pack_t<<<(256 * 576 + 255) / 256, 256, 0, stream>>>(t2_w, t2_b, wt2, tb2);
    qkv_wn<<<1, 192, 0, stream>>>(q_v, q_g, q_b, k_v, k_g, k_b, v_v, v_g, v_b, qkvw, qkvb);

    // f1/f2 = pc convs (Co=128 fused), f32 + f16 outputs
    conv3x3_igemm<<<dim3(128, 1), 256, 0, stream>>>(wpc, pcb, xh, kmeta,
                                                    f12, f12h, 128, 256, 0);

    // q,k,v 1x1 convs
    qkv_gemm<<<dim3(256, 3), 256, 0, stream>>>(qkvw, qkvb, f12h, qTb, kTb, vmb);

    // position attention, two-pass streaming softmax, all WMMA
    attn_rowstats<<<BATCH * 256, 256, 0, stream>>>(qTb, kTb, stb);
    attn_pos<<<BATCH * 128, 256, 0, stream>>>(qTb, kTb, vmb, stb, f12, gamma1, ph);

    // channel attention (tiny)
    ch_gram<<<BATCH * 64, 256, 0, stream>>>(f12, Gb);
    ch_softmax<<<BATCH * 64, 64, 0, stream>>>(Gb, attb);
    ch_apply<<<BATCH * 64 * 16, 256, 0, stream>>>(f12, attb, gamma2, chh);

    // out = t1(p) + t2(c)
    conv3x3_igemm<<<dim3(128, 2), 256, 0, stream>>>(wt1, tb1, ph, kmeta,
                                                    out, nullptr, 256, 64, 0);
    conv3x3_igemm<<<dim3(128, 2), 256, 0, stream>>>(wt2, tb2, chh, kmeta,
                                                    out, nullptr, 256, 64, 1);
}
